// FusionModel_61933428408555
// MI455X (gfx1250) — matrix operand order repaired and checked by reading, not compiled, not run on hardware
//
#include <hip/hip_runtime.h>
#include <hip/hip_bf16.h>
#include <math.h>

#define NN   8192
#define EE   262144
#define ETOT (EE + NN)
#define DIN  256
#define HIDC 64

typedef __attribute__((ext_vector_type(16))) _Float16 v16h;
typedef __attribute__((ext_vector_type(8)))  float    v8f;

#if __has_builtin(__builtin_amdgcn_exp2f)
#define EXP2F(x) __builtin_amdgcn_exp2f(x)
#else
#define EXP2F(x) exp2f(x)
#endif

__device__ __forceinline__ v8f wmma16(v16h a, v16h b, v8f c) {
  return __builtin_amdgcn_wmma_f32_16x16x32_f16(false, a, false, b, (short)0, c, false, false);
}
__device__ __forceinline__ v8f zero8() {
  v8f z;
#pragma unroll
  for (int i = 0; i < 8; ++i) z[i] = 0.0f;
  return z;
}
__device__ __forceinline__ unsigned int fenc(float f) {
  unsigned int u = __float_as_uint(f);
  return (u & 0x80000000u) ? ~u : (u | 0x80000000u);
}
__device__ __forceinline__ float fdec(unsigned int u) {
  return (u & 0x80000000u) ? __uint_as_float(u & 0x7FFFFFFFu) : __uint_as_float(~u);
}

// ---------------------------------------------------------------------------
// Generic WMMA GEMM: C[M,N] = act(A[M,K] @ B[K,N] + bias)
// ACT: 0=none 1=relu 2=elu.  K<=256, K%32==0, M%64==0, N%16==0.
// ---------------------------------------------------------------------------
template <int ACT>
__global__ __launch_bounds__(128) void wmma_gemm_kernel(
    const float* __restrict__ A, const float* __restrict__ B,
    const float* __restrict__ bias, float* __restrict__ C, int M, int N, int K) {
  __shared__ _Float16 BT[16][DIN + 8];
  __shared__ float biasS[16];
  const int tid  = threadIdx.x;
  const int lane = tid & 31;
  const int wave = tid >> 5;
  const int n0 = blockIdx.y * 16;
  const int m0 = blockIdx.x * 64 + wave * 16;

  for (int idx = tid; idx < 16 * K; idx += 128) {
    int col = idx / K, kk = idx - col * K;
    BT[col][kk] = (_Float16)B[(size_t)kk * N + n0 + col];
  }
  if (tid < 16) biasS[tid] = bias ? bias[n0 + tid] : 0.0f;
  __syncthreads();

  v8f acc = zero8();
  const int rowA = m0 + (lane & 15);
  const float* Arow = A + (size_t)rowA * K + ((lane < 16) ? 0 : 16);
  const int colB = lane & 15;
  const int kbB  = (lane < 16) ? 0 : 16;

  for (int kk = 0; kk < K; kk += 32) {
    const float* pa = Arow + kk;
    __builtin_prefetch(pa + 32, 0, 0);
    v16h a;
#pragma unroll
    for (int i = 0; i < 8; ++i) a[i] = (_Float16)pa[i];
#pragma unroll
    for (int i = 0; i < 8; ++i) a[8 + i] = (_Float16)pa[8 + i];
    v16h b;
    const _Float16* pb = &BT[colB][kk + kbB];
#pragma unroll
    for (int j = 0; j < 16; ++j) b[j] = pb[j];
    acc = wmma16(a, b, acc);
  }
#pragma unroll
  for (int r = 0; r < 8; ++r) {
    int rr = m0 + r + ((lane < 16) ? 0 : 8);
    float v = acc[r] + biasS[colB];
    if (ACT == 1) v = fmaxf(v, 0.0f);
    if (ACT == 2) v = v > 0.0f ? v : (__expf(v) - 1.0f);
    C[(size_t)rr * N + n0 + colB] = v;
  }
}

// ---------------------------------------------------------------------------
// Convert Q (pre-scaled by log2(e)/sqrt(32) so scores live in the exp2
// domain), K to f16 row-major, V to f16 transposed [64][NN]; all attention
// fragment loads become contiguous 16-half runs.
// ---------------------------------------------------------------------------
__global__ void qkv_cvt_kernel(const float* __restrict__ q,
                               const float* __restrict__ k,
                               const float* __restrict__ v,
                               _Float16* __restrict__ Qh,
                               _Float16* __restrict__ Kh,
                               _Float16* __restrict__ Vt) {
  int t = blockIdx.x * blockDim.x + threadIdx.x;
  if (t >= NN * 64) return;
  int n = t >> 6, c = t & 63;
  Qh[t] = (_Float16)(q[t] * 0.2550336690427058f);  // log2(e)/sqrt(32)
  Kh[t] = (_Float16)k[t];
  Vt[(size_t)c * NN + n] = (_Float16)v[t];
}

// ---------------------------------------------------------------------------
// Cross attention: hd=32, 2 heads. Block = 8 waves = 128 query rows; waves
// share 128-key f16 LDS chunks of K (and transposed V).
// Softmax is shift-invariant and scores here are small/bounded, so we skip
// the max pass: pass 1 accumulates l = sum(exp2(s)) per row (slot-partial,
// summed across the 16-lane row group); pass 2 recomputes S, writes the
// head-mean probabilities straight to d_out and accumulates O = P @ V.
// Both heads' WMMAs issue back-to-back so the second covers the WMMA->VALU
// hazard window of the first.
// ---------------------------------------------------------------------------
__global__ __launch_bounds__(256) void attn_kernel(
    const _Float16* __restrict__ Qh, const _Float16* __restrict__ Kh,
    const _Float16* __restrict__ Vt, float* __restrict__ attnW,
    float* __restrict__ Obuf) {
  __shared__ _Float16 Ksh[128 * 64];      // [key][dim]
  __shared__ _Float16 Vsh[64 * 128];      // [dim][key]
  __shared__ _Float16 pst[8][16 * 32];    // per-wave P transpose staging
  const int tid  = threadIdx.x;
  const int lane = tid & 31;
  const int w    = tid >> 5;
  const int hi16 = (lane < 16) ? 0 : 1;
  const int q0   = blockIdx.x * 128 + w * 16;

  // Q fragments (A-layout, already scaled into exp2 domain)
  v16h qa[2];
#pragma unroll
  for (int h = 0; h < 2; ++h) {
    const _Float16* pq = Qh + (size_t)(q0 + (lane & 15)) * 64 + 32 * h + (hi16 ? 16 : 0);
#pragma unroll
    for (int i = 0; i < 8; ++i) qa[h][i] = pq[i];
#pragma unroll
    for (int i = 0; i < 8; ++i) qa[h][8 + i] = pq[8 + i];
  }

  float lrow[2][8];
#pragma unroll
  for (int h = 0; h < 2; ++h)
#pragma unroll
    for (int r = 0; r < 8; ++r) lrow[h][r] = 0.0f;

  // ---- Pass 1: l = sum over keys of exp2(s) ----
  for (int ch = 0; ch < NN / 128; ++ch) {
    const uint4* src = (const uint4*)(Kh + (size_t)ch * 128 * 64);
    uint4* dst = (uint4*)Ksh;
#pragma unroll
    for (int j = 0; j < 4; ++j) dst[tid + 256 * j] = src[tid + 256 * j];
    __syncthreads();
#pragma unroll
    for (int tl = 0; tl < 8; ++tl) {
      const _Float16* pb0 = Ksh + (tl * 16 + (lane & 15)) * 64 + (hi16 ? 16 : 0);
      v16h b0, b1;
#pragma unroll
      for (int j = 0; j < 16; ++j) b0[j] = pb0[j];
#pragma unroll
      for (int j = 0; j < 16; ++j) b1[j] = pb0[32 + j];
      v8f s0 = wmma16(qa[0], b0, zero8());
      v8f s1 = wmma16(qa[1], b1, zero8());
#pragma unroll
      for (int r = 0; r < 8; ++r) lrow[0][r] += EXP2F(s0[r]);
#pragma unroll
      for (int r = 0; r < 8; ++r) lrow[1][r] += EXP2F(s1[r]);
    }
    __syncthreads();
  }
  // sum partials across the 16 lanes sharing each row; fold 1/l
#pragma unroll
  for (int h = 0; h < 2; ++h)
#pragma unroll
    for (int r = 0; r < 8; ++r) {
      float l = lrow[h][r];
      for (int off = 1; off < 16; off <<= 1) l += __shfl_xor(l, off, 32);
      lrow[h][r] = 1.0f / l;
    }

  // ---- Pass 2: probabilities out + O = P @ V ----
  v8f o[2][2];
#pragma unroll
  for (int h = 0; h < 2; ++h)
#pragma unroll
    for (int g = 0; g < 2; ++g) o[h][g] = zero8();

  for (int ch = 0; ch < NN / 128; ++ch) {
    {
      const uint4* src = (const uint4*)(Kh + (size_t)ch * 128 * 64);
      uint4* dst = (uint4*)Ksh;
#pragma unroll
      for (int j = 0; j < 4; ++j) dst[tid + 256 * j] = src[tid + 256 * j];
    }
    for (int i = tid; i < 512; i += 256) {
      int row = i >> 3, seg = i & 7;
      ((uint4*)(Vsh + row * 128))[seg] =
          ((const uint4*)(Vt + (size_t)row * NN + ch * 128))[seg];
    }
    __syncthreads();

    for (int p = 0; p < 4; ++p) {  // 32-key groups in chunk
      float pv[2][2][8];           // [head][sub][slot]
#pragma unroll
      for (int sub = 0; sub < 2; ++sub) {
        const _Float16* pb =
            Ksh + ((p * 2 + sub) * 16 + (lane & 15)) * 64 + (hi16 ? 16 : 0);
        v16h b0, b1;
#pragma unroll
        for (int j = 0; j < 16; ++j) b0[j] = pb[j];
#pragma unroll
        for (int j = 0; j < 16; ++j) b1[j] = pb[32 + j];
        v8f s0 = wmma16(qa[0], b0, zero8());
        v8f s1 = wmma16(qa[1], b1, zero8());
#pragma unroll
        for (int r = 0; r < 8; ++r) pv[0][sub][r] = EXP2F(s0[r]) * lrow[0][r];
#pragma unroll
        for (int r = 0; r < 8; ++r) pv[1][sub][r] = EXP2F(s1[r]) * lrow[1][r];
      }
      // head-averaged weights -> d_out (each (q,k) written exactly once)
#pragma unroll
      for (int sub = 0; sub < 2; ++sub) {
        int col = ch * 128 + p * 32 + sub * 16 + (lane & 15);
#pragma unroll
        for (int r = 0; r < 8; ++r) {
          int row = q0 + r + (hi16 ? 8 : 0);
          attnW[(size_t)row * NN + col] = 0.5f * (pv[0][sub][r] + pv[1][sub][r]);
        }
      }
      // P (C-layout) -> wave-private LDS -> A-layout f16, O += P @ V
      for (int h = 0; h < 2; ++h) {
#pragma unroll
        for (int sub = 0; sub < 2; ++sub) {
          int col = sub * 16 + (lane & 15);
#pragma unroll
          for (int r = 0; r < 8; ++r)
            pst[w][(r + (hi16 ? 8 : 0)) * 32 + col] = (_Float16)pv[h][sub][r];
        }
        v16h ap;
        {
          const _Float16* pp = &pst[w][(lane & 15) * 32];
          int cb = hi16 ? 16 : 0;
#pragma unroll
          for (int i = 0; i < 8; ++i) ap[i] = pp[cb + i];
#pragma unroll
          for (int i = 0; i < 8; ++i) ap[8 + i] = pp[8 + cb + i];
        }
#pragma unroll
        for (int g = 0; g < 2; ++g) {
          int cd = g * 16 + (lane & 15);
          const _Float16* pvv = Vsh + cd * 128 + p * 32 + (hi16 ? 16 : 0);
          v16h vb;
#pragma unroll
          for (int j = 0; j < 16; ++j) vb[j] = pvv[j];
          o[h][g] = wmma16(ap, vb, o[h][g]);
        }
      }
    }
    __syncthreads();
  }

#pragma unroll
  for (int h = 0; h < 2; ++h)
#pragma unroll
    for (int g = 0; g < 2; ++g)
#pragma unroll
      for (int r = 0; r < 8; ++r) {
        int row = q0 + r + (hi16 ? 8 : 0);
        int col = 32 * h + g * 16 + (lane & 15);
        Obuf[(size_t)row * 64 + col] = o[h][g][r];
      }
}

// ---------------------------------------------------------------------------
// GAT / GCN helper kernels (atomics, bandwidth-bound)
// ---------------------------------------------------------------------------
__global__ void fill_u32_kernel(unsigned int* p, unsigned int v, int n) {
  int t = blockIdx.x * blockDim.x + threadIdx.x;
  if (t < n) p[t] = v;
}

__global__ void gat_alpha_kernel(const float* __restrict__ h,
                                 const float* __restrict__ att_src,
                                 const float* __restrict__ att_dst,
                                 float* a_s, float* a_d, int H, int C) {
  int t = blockIdx.x * blockDim.x + threadIdx.x;
  if (t >= NN * H) return;
  int n = t / H, hh = t - n * H;
  const float* hp = h + (size_t)n * H * C + hh * C;
  float ss = 0.f, dd = 0.f;
  for (int c = 0; c < C; ++c) {
    float v = hp[c];
    ss += v * att_src[hh * C + c];
    dd += v * att_dst[hh * C + c];
  }
  a_s[t] = ss; a_d[t] = dd;
}

__global__ void gat_edge_max_kernel(const int* __restrict__ ei,
                                    const float* __restrict__ a_s,
                                    const float* __restrict__ a_d,
                                    unsigned int* emax, int H) {
  int t = blockIdx.x * blockDim.x + threadIdx.x;
  if (t >= ETOT * H) return;
  int e = t / H, hh = t - e * H;
  int s = e < EE ? ei[e] : (e - EE);
  int d = e < EE ? ei[EE + e] : (e - EE);
  float v = a_s[s * H + hh] + a_d[d * H + hh];
  v = v > 0.f ? v : 0.2f * v;
  atomicMax(&emax[d * H + hh], fenc(v));
}

__global__ void gat_edge_w_kernel(const int* __restrict__ ei,
                                  const float* __restrict__ a_s,
                                  const float* __restrict__ a_d,
                                  const unsigned int* __restrict__ emax,
                                  float* denom, float* wedge, int H) {
  int t = blockIdx.x * blockDim.x + threadIdx.x;
  if (t >= ETOT * H) return;
  int e = t / H, hh = t - e * H;
  int s = e < EE ? ei[e] : (e - EE);
  int d = e < EE ? ei[EE + e] : (e - EE);
  float v = a_s[s * H + hh] + a_d[d * H + hh];
  v = v > 0.f ? v : 0.2f * v;
  float ww = __expf(v - fdec(emax[d * H + hh]));
  wedge[t] = ww;
  atomicAdd(&denom[d * H + hh], ww);
}

__global__ void gat_edge_acc_kernel(const int* __restrict__ ei,
                                    const float* __restrict__ h,
                                    const float* __restrict__ wedge,
                                    float* numer, int H, int C) {
  int t = blockIdx.x * blockDim.x + threadIdx.x;
  int HC = H * C;
  if (t >= ETOT * HC) return;
  int e = t / HC, c = t - e * HC, hh = c / C;
  int s = e < EE ? ei[e] : (e - EE);
  int d = e < EE ? ei[EE + e] : (e - EE);
  atomicAdd(&numer[(size_t)d * HC + c], h[(size_t)s * HC + c] * wedge[e * H + hh]);
}

__global__ void gat_node_kernel(const float* __restrict__ numer,
                                const float* __restrict__ denom,
                                const float* __restrict__ b, float* out,
                                int H, int C) {
  int t = blockIdx.x * blockDim.x + threadIdx.x;
  int HC = H * C;
  if (t >= NN * HC) return;
  int n = t / HC, c = t - n * HC, hh = c / C;
  float v = numer[t] / (denom[n * H + hh] + 1e-16f) + b[c];
  out[t] = v > 0.f ? v : (__expf(v) - 1.0f);  // ELU
}

__global__ void deg_kernel(const int* __restrict__ ei, float* deg) {
  int e = blockIdx.x * blockDim.x + threadIdx.x;
  if (e >= ETOT) return;
  int d = e < EE ? ei[EE + e] : (e - EE);
  atomicAdd(&deg[d], 1.0f);
}

__global__ void dinv_kernel(const float* __restrict__ deg, float* dinv) {
  int n = blockIdx.x * blockDim.x + threadIdx.x;
  if (n < NN) dinv[n] = rsqrtf(fmaxf(deg[n], 1.0f));
}

__global__ void gcn_edge_kernel(const int* __restrict__ ei,
                                const float* __restrict__ g,
                                const float* __restrict__ dinv, float* acc) {
  int t = blockIdx.x * blockDim.x + threadIdx.x;
  if (t >= ETOT * HIDC) return;
  int e = t / HIDC, c = t - e * HIDC;
  int s = e < EE ? ei[e] : (e - EE);
  int d = e < EE ? ei[EE + e] : (e - EE);
  atomicAdd(&acc[(size_t)d * HIDC + c], g[(size_t)s * HIDC + c] * dinv[s] * dinv[d]);
}

__global__ void gcn_node_kernel(const float* __restrict__ acc,
                                const float* __restrict__ b, float* out) {
  int t = blockIdx.x * blockDim.x + threadIdx.x;
  if (t >= NN * HIDC) return;
  int c = t % HIDC;
  out[t] = fmaxf(acc[t] + b[c], 0.0f);
}

__global__ void fuse_kernel(const float* __restrict__ a,
                            const float* __restrict__ bsrc, float* out) {
  int t = blockIdx.x * blockDim.x + threadIdx.x;
  if (t >= NN * 128) return;
  int n = t >> 7, c = t & 127;
  out[t] = (c < 64) ? a[(size_t)n * 64 + c] : bsrc[(size_t)n * 64 + (c - 64)];
}

__global__ void pred_kernel(const float* __restrict__ hid,
                            const float* __restrict__ w2,
                            const float* __restrict__ b2, float* pred) {
  int n = blockIdx.x * blockDim.x + threadIdx.x;
  if (n >= NN) return;
  const float* hp = hid + (size_t)n * 64;
  float s = 0.f;
#pragma unroll
  for (int j = 0; j < 64; ++j) s += hp[j] * w2[j];
  pred[n] = s + b2[0];
}

// ---------------------------------------------------------------------------
extern "C" void kernel_launch(void* const* d_in, const int* in_sizes, int n_in,
                              void* d_out, int out_size, void* d_ws, size_t ws_size,
                              hipStream_t stream) {
  const float* x      = (const float*)d_in[0];
  const int*   ei_t   = (const int*)d_in[1];
  const int*   ei_e   = (const int*)d_in[2];
  const float* gat1_W = (const float*)d_in[3];
  const float* g1as   = (const float*)d_in[4];
  const float* g1ad   = (const float*)d_in[5];
  const float* gat1_b = (const float*)d_in[6];
  const float* gat2_W = (const float*)d_in[7];
  const float* g2as   = (const float*)d_in[8];
  const float* g2ad   = (const float*)d_in[9];
  const float* gat2_b = (const float*)d_in[10];
  const float* gcn1_W = (const float*)d_in[11];
  const float* gcn1_b = (const float*)d_in[12];
  const float* gcn2_W = (const float*)d_in[13];
  const float* gcn2_b = (const float*)d_in[14];
  const float* Wq = (const float*)d_in[15]; const float* bq = (const float*)d_in[16];
  const float* Wk = (const float*)d_in[17]; const float* bk = (const float*)d_in[18];
  const float* Wv = (const float*)d_in[19]; const float* bv = (const float*)d_in[20];
  const float* Wo = (const float*)d_in[21]; const float* bo = (const float*)d_in[22];
  const float* fc1_W = (const float*)d_in[23]; const float* fc1_b = (const float*)d_in[24];
  const float* fc2_W = (const float*)d_in[25]; const float* fc2_b = (const float*)d_in[26];

  float* pred  = (float*)d_out;
  float* attnW = (float*)d_out + NN;

  float* base = (float*)d_ws;
  size_t off = 0;
  auto alloc = [&](size_t n) { float* r = base + off; off += n; return r; };

  float* h1     = alloc((size_t)NN * 128);
  float* as1    = alloc((size_t)NN * 2);
  float* ad1    = alloc((size_t)NN * 2);
  unsigned int* emax1 = (unsigned int*)alloc((size_t)NN * 2);
  float* den1   = alloc((size_t)NN * 2);
  float* wed1   = alloc((size_t)ETOT * 2);
  float* num1   = alloc((size_t)NN * 128);
  float* htkg1  = alloc((size_t)NN * 128);
  float* h2lin  = alloc((size_t)NN * 64);
  float* as2    = alloc((size_t)NN);
  float* ad2    = alloc((size_t)NN);
  unsigned int* emax2 = (unsigned int*)alloc((size_t)NN);
  float* den2   = alloc((size_t)NN);
  float* wed2   = alloc((size_t)ETOT);
  float* num2   = alloc((size_t)NN * 64);
  float* htkg2  = alloc((size_t)NN * 64);
  float* glin1  = alloc((size_t)NN * 64);
  float* deg    = alloc((size_t)NN);
  float* dinv   = alloc((size_t)NN);
  float* acc1   = alloc((size_t)NN * 64);
  float* hekg1  = alloc((size_t)NN * 64);
  float* glin2  = alloc((size_t)NN * 64);
  float* acc2   = alloc((size_t)NN * 64);
  float* hekg2  = alloc((size_t)NN * 64);
  float* qb     = alloc((size_t)NN * 64);
  float* kb     = alloc((size_t)NN * 64);
  float* vb     = alloc((size_t)NN * 64);
  float* Obuf   = alloc((size_t)NN * 64);
  float* aproj  = alloc((size_t)NN * 64);
  float* fused  = alloc((size_t)NN * 128);
  float* hid1   = alloc((size_t)NN * 64);
  _Float16* Qh  = (_Float16*)alloc((size_t)NN * 32);
  _Float16* Kh  = (_Float16*)alloc((size_t)NN * 32);
  _Float16* Vt  = (_Float16*)alloc((size_t)NN * 32);

  const unsigned int ENC_NEG_INF = 0x007FFFFFu;
  auto cdiv = [](int a, int b) { return (a + b - 1) / b; };

  // ---- GAT layer 1 (H=2, C=64) ----
  wmma_gemm_kernel<0><<<dim3(NN / 64, 128 / 16), 128, 0, stream>>>(x, gat1_W, nullptr, h1, NN, 128, DIN);
  gat_alpha_kernel<<<cdiv(NN * 2, 256), 256, 0, stream>>>(h1, g1as, g1ad, as1, ad1, 2, 64);
  fill_u32_kernel<<<cdiv(NN * 2, 256), 256, 0, stream>>>(emax1, ENC_NEG_INF, NN * 2);
  hipMemsetAsync(den1, 0, (size_t)NN * 2 * sizeof(float), stream);
  hipMemsetAsync(num1, 0, (size_t)NN * 128 * sizeof(float), stream);
  gat_edge_max_kernel<<<cdiv(ETOT * 2, 256), 256, 0, stream>>>(ei_t, as1, ad1, emax1, 2);
  gat_edge_w_kernel<<<cdiv(ETOT * 2, 256), 256, 0, stream>>>(ei_t, as1, ad1, emax1, den1, wed1, 2);
  gat_edge_acc_kernel<<<cdiv(ETOT * 128, 256), 256, 0, stream>>>(ei_t, h1, wed1, num1, 2, 64);
  gat_node_kernel<<<cdiv(NN * 128, 256), 256, 0, stream>>>(num1, den1, gat1_b, htkg1, 2, 64);

  // ---- GAT layer 2 (H=1, C=64) ----
  wmma_gemm_kernel<0><<<dim3(NN / 64, 64 / 16), 128, 0, stream>>>(htkg1, gat2_W, nullptr, h2lin, NN, 64, 128);
  gat_alpha_kernel<<<cdiv(NN, 256), 256, 0, stream>>>(h2lin, g2as, g2ad, as2, ad2, 1, 64);
  fill_u32_kernel<<<cdiv(NN, 256), 256, 0, stream>>>(emax2, ENC_NEG_INF, NN);
  hipMemsetAsync(den2, 0, (size_t)NN * sizeof(float), stream);
  hipMemsetAsync(num2, 0, (size_t)NN * 64 * sizeof(float), stream);
  gat_edge_max_kernel<<<cdiv(ETOT, 256), 256, 0, stream>>>(ei_t, as2, ad2, emax2, 1);
  gat_edge_w_kernel<<<cdiv(ETOT, 256), 256, 0, stream>>>(ei_t, as2, ad2, emax2, den2, wed2, 1);
  gat_edge_acc_kernel<<<cdiv(ETOT * 64, 256), 256, 0, stream>>>(ei_t, h2lin, wed2, num2, 1, 64);
  gat_node_kernel<<<cdiv(NN * 64, 256), 256, 0, stream>>>(num2, den2, gat2_b, htkg2, 1, 64);

  // ---- GCN branch ----
  wmma_gemm_kernel<0><<<dim3(NN / 64, 64 / 16), 128, 0, stream>>>(x, gcn1_W, nullptr, glin1, NN, 64, DIN);
  hipMemsetAsync(deg, 0, (size_t)NN * sizeof(float), stream);
  deg_kernel<<<cdiv(ETOT, 256), 256, 0, stream>>>(ei_e, deg);
  dinv_kernel<<<cdiv(NN, 256), 256, 0, stream>>>(deg, dinv);
  hipMemsetAsync(acc1, 0, (size_t)NN * 64 * sizeof(float), stream);
  gcn_edge_kernel<<<cdiv(ETOT * 64, 256), 256, 0, stream>>>(ei_e, glin1, dinv, acc1);
  gcn_node_kernel<<<cdiv(NN * 64, 256), 256, 0, stream>>>(acc1, gcn1_b, hekg1);
  wmma_gemm_kernel<0><<<dim3(NN / 64, 64 / 16), 128, 0, stream>>>(hekg1, gcn2_W, nullptr, glin2, NN, 64, 64);
  hipMemsetAsync(acc2, 0, (size_t)NN * 64 * sizeof(float), stream);
  gcn_edge_kernel<<<cdiv(ETOT * 64, 256), 256, 0, stream>>>(ei_e, glin2, dinv, acc2);
  gcn_node_kernel<<<cdiv(NN * 64, 256), 256, 0, stream>>>(acc2, gcn2_b, hekg2);

  // ---- Cross attention ----
  wmma_gemm_kernel<0><<<dim3(NN / 64, 64 / 16), 128, 0, stream>>>(htkg2, Wq, bq, qb, NN, 64, 64);
  wmma_gemm_kernel<0><<<dim3(NN / 64, 64 / 16), 128, 0, stream>>>(hekg2, Wk, bk, kb, NN, 64, 64);
  wmma_gemm_kernel<0><<<dim3(NN / 64, 64 / 16), 128, 0, stream>>>(hekg2, Wv, bv, vb, NN, 64, 64);
  qkv_cvt_kernel<<<cdiv(NN * 64, 256), 256, 0, stream>>>(qb, kb, vb, Qh, Kh, Vt);
  attn_kernel<<<NN / 128, 256, 0, stream>>>(Qh, Kh, Vt, attnW, Obuf);
  wmma_gemm_kernel<0><<<dim3(NN / 64, 64 / 16), 128, 0, stream>>>(Obuf, Wo, bo, aproj, NN, 64, 64);

  // ---- Head ----
  fuse_kernel<<<cdiv(NN * 128, 256), 256, 0, stream>>>(htkg2, aproj, fused);
  wmma_gemm_kernel<1><<<dim3(NN / 64, 64 / 16), 128, 0, stream>>>(fused, fc1_W, fc1_b, hid1, NN, 64, 128);
  pred_kernel<<<cdiv(NN, 256), 256, 0, stream>>>(hid1, fc2_W, fc2_b, pred);
}
